// mDACF_33371895889930
// MI455X (gfx1250) — matrix operand order repaired and checked by reading, not compiled, not run on hardware
//
#include <hip/hip_runtime.h>
#include <hip/hip_bf16.h>
#include <stdint.h>

typedef __attribute__((ext_vector_type(16))) __bf16 v16bf;
typedef __attribute__((ext_vector_type(8)))  float  v8f;

#define USER_DIM 8000
#define XDIM     16000
#define LATENT   512
#define BATCH    4096
#define KDIM     8000

#define BM 128   // workgroup M tile
#define BN 64    // workgroup N tile
#define BK 64    // K panel per barrier

__device__ __forceinline__ uint32_t f2bf(float f) {
  // round-to-nearest-even fp32 -> bf16 (returned in low 16 bits)
  uint32_t u = __float_as_uint(f);
  return (u + 0x7FFFu + ((u >> 16) & 1u)) >> 16;
}

// W [K=8000][N=512] f32 -> Wt [N=512][K=8000] bf16, via LDS tile transpose so
// both global reads (float4) and global writes (uint4) are coalesced.
__global__ __launch_bounds__(256) void convert_w(const float* __restrict__ W,
                                                 uint16_t* __restrict__ Wt) {
  __shared__ uint16_t tile[64][72];  // 72-u16 pitch: rows 16B-aligned, conflict-light
  const int t  = threadIdx.x;
  const int kt = blockIdx.x * 64;  // 125 tiles along K
  const int nt = blockIdx.y * 64;  // 8 tiles along N

  // read 64(k) x 64(n) f32 tile, coalesced float4; scatter bf16 into transposed LDS
#pragma unroll
  for (int i = 0; i < 4; ++i) {
    int e   = t + 256 * i;  // 0..1023 float4s
    int row = e >> 4;       // k-local, 16 float4 per row
    int c4  = e & 15;       // n-local / 4
    const float4 v = *(const float4*)(W + (size_t)(kt + row) * LATENT + nt + c4 * 4);
    tile[c4 * 4 + 0][row] = (uint16_t)f2bf(v.x);
    tile[c4 * 4 + 1][row] = (uint16_t)f2bf(v.y);
    tile[c4 * 4 + 2][row] = (uint16_t)f2bf(v.z);
    tile[c4 * 4 + 3][row] = (uint16_t)f2bf(v.w);
  }
  __syncthreads();

  // write 64(n) x 64(k) bf16 tile, coalesced uint4 (8 bf16 each)
#pragma unroll
  for (int i = 0; i < 2; ++i) {
    int e   = t + 256 * i;  // 0..511 uint4s
    int row = e >> 3;       // n-local, 8 uint4 per row
    int c8  = e & 7;        // k-local / 8
    uint4 p = *(const uint4*)&tile[row][c8 * 8];
    *(uint4*)(Wt + (size_t)(nt + row) * KDIM + kt + c8 * 8) = p;
  }
}

union FragAB { uint4 q[2]; v16bf v; };

#define WMMA_BF16(A, B, C) \
  __builtin_amdgcn_wmma_f32_16x16x32_bf16(false, (A), false, (B), (short)0, (C), false, false)

// Z[4096][512] = bf16(X[:, colOff:colOff+8000]) @ Wt^T + bias
// block = 256 threads (8 waves); WG tile 128x64; each wave owns 32x32 (2x2 WMMA tiles)
__global__ __launch_bounds__(256) void gemm_enc(const float* __restrict__ X, int colOff,
                                                const uint16_t* __restrict__ Wt,
                                                const float* __restrict__ bias,
                                                float* __restrict__ Z) {
  __shared__ uint16_t As[BM * BK];  // [m][k] bf16, 16 KB
  __shared__ uint16_t Bs[BN * BK];  // [n][k] bf16,  8 KB

  const int t    = threadIdx.x;
  const int lane = t & 31;
  const int w    = t >> 5;
  const int h    = lane >> 4;   // lane half per ISA WMMA layouts
  const int ln   = lane & 15;
  const int mt   = w & 3;       // 4 M-groups of 32 rows
  const int ng   = w >> 2;      // 2 N-groups of 32 cols
  const int m0   = blockIdx.y * BM;
  const int n0   = blockIdx.x * BN;

  v8f acc00 = {}, acc01 = {}, acc10 = {}, acc11 = {};

  for (int k0 = 0; k0 < KDIM; k0 += BK) {
    // prefetch next K panel into GL2 while this one is staged/consumed
    if (k0 + BK < KDIM) {
      __builtin_prefetch(X + (size_t)(m0 + (t >> 1)) * XDIM + colOff + k0 + BK + (t & 1) * 32, 0, 0);
      __builtin_prefetch(Wt + (size_t)(n0 + (t & 63)) * KDIM + k0 + BK + (t >> 6) * 16, 0, 0);
    }
    // stage A: 128x64 fp32 -> bf16 (2048 float4 loads -> ds_store_b64)
#pragma unroll
    for (int i = 0; i < 8; ++i) {
      int e   = t + 256 * i;        // 0..2047
      int row = e >> 4;             // 16 float4 per row
      int c4  = e & 15;
      const float4 xv =
          *(const float4*)(X + (size_t)(m0 + row) * XDIM + colOff + k0 + c4 * 4);
      uint2 pp;
      pp.x = f2bf(xv.x) | (f2bf(xv.y) << 16);
      pp.y = f2bf(xv.z) | (f2bf(xv.w) << 16);
      *(uint2*)&As[row * BK + c4 * 4] = pp;
    }
    // stage B: 64(n) x 64(k) bf16 rows of Wt (512 uint4 copies, b128 both ways)
#pragma unroll
    for (int i = 0; i < 2; ++i) {
      int e   = t + 256 * i;        // 0..511
      int row = e >> 3;             // 8 uint4 per row
      int c8  = e & 7;
      uint4 p = *(const uint4*)(Wt + (size_t)(n0 + row) * KDIM + k0 + c8 * 8);
      *(uint4*)&Bs[row * BK + c8 * 8] = p;
    }
    __syncthreads();

#pragma unroll
    for (int ks = 0; ks < BK; ks += 32) {
      FragAB a0, a1, b0, b1;
      // A fragments (16x32 bf16): lane row M = ln; halves at K = ks+h*8, ks+16+h*8
      const int am0 = (mt * 32 + ln) * BK + ks;
      a0.q[0] = *(const uint4*)&As[am0 + h * 8];
      a0.q[1] = *(const uint4*)&As[am0 + 16 + h * 8];
      const int am1 = am0 + 16 * BK;
      a1.q[0] = *(const uint4*)&As[am1 + h * 8];
      a1.q[1] = *(const uint4*)&As[am1 + 16 + h * 8];
      // B fragments (32x16 bf16): lane col N = ln; halves at K = ks+h*8, ks+16+h*8
      const int bn0 = (ng * 32 + ln) * BK + ks;
      b0.q[0] = *(const uint4*)&Bs[bn0 + h * 8];
      b0.q[1] = *(const uint4*)&Bs[bn0 + 16 + h * 8];
      const int bn1 = bn0 + 16 * BK;
      b1.q[0] = *(const uint4*)&Bs[bn1 + h * 8];
      b1.q[1] = *(const uint4*)&Bs[bn1 + 16 + h * 8];

      acc00 = WMMA_BF16(a0.v, b0.v, acc00);
      acc01 = WMMA_BF16(a0.v, b1.v, acc01);
      acc10 = WMMA_BF16(a1.v, b0.v, acc10);
      acc11 = WMMA_BF16(a1.v, b1.v, acc11);
    }
    __syncthreads();
  }

  // epilogue: C layout -> M = r + 8*h, N = ln ; add encoder bias
  const int colA = n0 + ng * 32 + ln;
  const int colB = colA + 16;
  const float bvA = bias[colA];
  const float bvB = bias[colB];
  const int row0 = m0 + mt * 32 + 8 * h;
  const int row1 = row0 + 16;
#pragma unroll
  for (int r = 0; r < 8; ++r) {
    Z[(size_t)(row0 + r) * LATENT + colA] = acc00[r] + bvA;
    Z[(size_t)(row0 + r) * LATENT + colB] = acc01[r] + bvB;
    Z[(size_t)(row1 + r) * LATENT + colA] = acc10[r] + bvA;
    Z[(size_t)(row1 + r) * LATENT + colB] = acc11[r] + bvB;
  }
}

// out[b] = dot(ZU[b,:], ZI[b,:]) — one wave32 per row, float4 coalesced loads
__global__ __launch_bounds__(256) void dot_diag(const float* __restrict__ ZU,
                                                const float* __restrict__ ZI,
                                                float* __restrict__ out) {
  const int lane = threadIdx.x & 31;
  const int w    = threadIdx.x >> 5;
  const int row  = blockIdx.x * 8 + w;
  const float* zu = ZU + (size_t)row * LATENT;
  const float* zi = ZI + (size_t)row * LATENT;
  float s = 0.f;
#pragma unroll
  for (int i = 0; i < LATENT / 128; ++i) {
    const float4 a = *(const float4*)(zu + lane * 4 + 128 * i);
    const float4 b = *(const float4*)(zi + lane * 4 + 128 * i);
    s += a.x * b.x + a.y * b.y + a.z * b.z + a.w * b.w;
  }
#pragma unroll
  for (int off = 16; off > 0; off >>= 1)
    s += __shfl_xor(s, off, 32);
  if (lane == 0) out[row] = s;
}

extern "C" void kernel_launch(void* const* d_in, const int* in_sizes, int n_in,
                              void* d_out, int out_size, void* d_ws, size_t ws_size,
                              hipStream_t stream) {
  // setup_inputs order: x, user_W_enc, user_b_enc, user_W_dec, user_b_dec,
  //                     item_W_enc, item_b_enc, item_W_dec, item_b_dec
  // (W_dec / b_dec are dead in the reference output.)
  const float* x   = (const float*)d_in[0];
  const float* uWe = (const float*)d_in[1];
  const float* ube = (const float*)d_in[2];
  const float* iWe = (const float*)d_in[5];
  const float* ibe = (const float*)d_in[6];
  float* out = (float*)d_out;

  char* ws = (char*)d_ws;
  const size_t szWt = (size_t)LATENT * KDIM * sizeof(uint16_t);  // 8,192,000 (256B-multiple)
  const size_t szZ  = (size_t)BATCH * LATENT * sizeof(float);    // 8,388,608
  uint16_t* WtU = (uint16_t*)(ws);
  uint16_t* WtI = (uint16_t*)(ws + szWt);
  float*    ZU  = (float*)(ws + 2 * szWt);
  float*    ZI  = (float*)(ws + 2 * szWt + szZ);

  dim3 cgrid(KDIM / 64, LATENT / 64);  // (125, 8)
  convert_w<<<cgrid, 256, 0, stream>>>(uWe, WtU);
  convert_w<<<cgrid, 256, 0, stream>>>(iWe, WtI);

  dim3 grid(LATENT / BN, BATCH / BM);  // (8, 32)
  gemm_enc<<<grid, 256, 0, stream>>>(x, 0,        WtU, ube, ZU);
  gemm_enc<<<grid, 256, 0, stream>>>(x, USER_DIM, WtI, ibe, ZI);

  dot_diag<<<BATCH / 8, 256, 0, stream>>>(ZU, ZI, out);
}
